// Moon_46746424049777
// MI455X (gfx1250) — matrix operand order repaired and checked by reading, not compiled, not run on hardware
//
#include <hip/hip_runtime.h>
#include <hip/hip_bf16.h>
#include <math.h>

typedef _Float16 half_t;
typedef __attribute__((ext_vector_type(16))) _Float16 v16h;
typedef __attribute__((ext_vector_type(8)))  _Float16 v8h;
typedef __attribute__((ext_vector_type(8)))  float    v8f;

#define N_ELEC    4096
#define N_NUC     512
#define N_EN      131072
#define N_EE      131072
#define N_EE_SAME 65536
#define N_NN      16384
#define EMB       256
#define INV_SQRT2 0.70710678118654752f

// ---------------------------------------------------------------- fast math

// Branch-free tanh: tanh(x) = sign(x) * (1 - 2/(exp(2|x|)+1)).
// Uses v_exp_f32 + v_rcp_f32; no EXEC divergence, ~1e-6 accuracy (far below
// the f16 operand quantization used in the WMMA path).
__device__ __forceinline__ float fast_tanh(float x) {
  float ax = fabsf(x);
  float e = __expf(2.0f * ax);
  float t = 1.0f - 2.0f / (e + 1.0f);
  return copysignf(t, x);
}

// Hardware fp32 atomic add (global_atomic_add_f32), avoids CAS-loop lowering.
__device__ __forceinline__ void atomic_add_f32(float* p, float v) {
  unsafeAtomicAdd(p, v);
}

// ---------------------------------------------------------------- WMMA helpers

__device__ __forceinline__ v8f wmma16(v16h a, v16h b, v8f c) {
  return __builtin_amdgcn_wmma_f32_16x16x32_f16(false, a, false, b, (short)0, c,
                                                false, false);
}

__device__ __forceinline__ v8f zero8() {
  v8f z = {0.f, 0.f, 0.f, 0.f, 0.f, 0.f, 0.f, 0.f};
  return z;
}

// A fragment (16xK tile, row-major, f16, leading dim ld in elements).
// Layout (ISA 7.12.2, 16-bit A 16x32): lanes 0-15 hold K=[0..7,16..23],
// lanes 16-31 hold K=[8..15,24..31] -> two contiguous 8-half runs per lane.
__device__ __forceinline__ v16h load_a16(const half_t* __restrict__ p, int ld) {
  const int lane = threadIdx.x & 31;
  const int m = lane & 15;
  const int h8 = (lane >> 4) << 3;             // 0 or 8
  const half_t* q = p + m * ld + h8;
  v8h lo = *reinterpret_cast<const v8h*>(q);
  v8h hi = *reinterpret_cast<const v8h*>(q + 16);
  v16h a;
#pragma unroll
  for (int i = 0; i < 8; ++i) { a[i] = lo[i]; a[i + 8] = hi[i]; }
  return a;
}

// B fragment from pre-packed weights: packed[(kt*NT+nt)*512 + lane*16 + h]
// holds W[kt*32 + (h&7) + (lane>=16)*8 + (h>>3)*16][nt*16 + (lane&15)] (B layout: lane=N col,
// lanes 0-15 K 0-7,16-23, lanes 16-31 K 8-15,24-31).
__device__ __forceinline__ v16h load_bp(const half_t* __restrict__ p, int NT,
                                        int kt, int nt) {
  const int lane = threadIdx.x & 31;
  return *reinterpret_cast<const v16h*>(p + (((kt * NT + nt) << 9) + (lane << 4)));
}

// ---------------------------------------------------------------- weight pack

__global__ void pack_b_kernel(const float* __restrict__ W, half_t* __restrict__ out,
                              int K, int N, int total) {
  int idx = blockIdx.x * 256 + threadIdx.x;
  if (idx >= total) return;
  int h = idx & 15;
  int lane = (idx >> 4) & 31;
  int tile = idx >> 9;
  int NT = N >> 4;
  int nt = tile % NT;
  int kt = tile / NT;
  int k = kt * 32 + (h & 7) + ((lane >> 4) << 3) + ((h >> 3) << 4);
  int n = nt * 16 + (lane & 15);
  out[idx] = (k < K) ? (half_t)W[k * N + n] : (half_t)0.f;
}

// ---------------------------------------------------------------- ee edges

__global__ __launch_bounds__(128) void ee_kernel(
    const float* __restrict__ ee_dists, const int* __restrict__ ee_i,
    const half_t* __restrict__ pW1s, const half_t* __restrict__ pW1d,
    const half_t* __restrict__ pW2s, const half_t* __restrict__ pW2d,
    const float* __restrict__ b1s, const float* __restrict__ b1d,
    const float* __restrict__ b2s, const float* __restrict__ b2d,
    const float* __restrict__ W_eedata, const float* __restrict__ b_eedata,
    float* __restrict__ e_emb) {
  __shared__ __align__(16) half_t feats[16 * 32];
  __shared__ float dvec[16 * 4];
  __shared__ int seg[16];
  __shared__ __align__(16) half_t H[16 * 256];

  const int tid = threadIdx.x;
  const int base = blockIdx.x * 16;
  const bool same = base < N_EE_SAME;

  if (tid < 16) {
    int e = base + tid;
    seg[tid] = ee_i[e] + (e >= N_EE_SAME ? N_ELEC : 0);
  }
  if (tid < 64) {
    int m = tid >> 2, c = tid & 3;
    int e = base + m;
    float r = ee_dists[e * 4 + 3];
    dvec[tid] = ee_dists[e * 4 + c] * (log1pf(r) / r);
  }
#pragma unroll
  for (int i = 0; i < 4; ++i) {
    int s = tid + i * 128;
    int m = s >> 5, k = s & 31;
    float r = ee_dists[(base + m) * 4 + 3];
    feats[s] = (k < 16) ? (half_t)__expf(-r * (float)(k + 1)) : (half_t)0.f;
  }
  __syncthreads();

  const int w = tid >> 5, lane = tid & 31;
  const int ncol = lane & 15, mb = (lane >> 4) << 3;
  const half_t* pW1 = same ? pW1s : pW1d;
  const half_t* pW2 = same ? pW2s : pW2d;
  const float* b1 = same ? b1s : b1d;
  const float* b2 = same ? b2s : b2d;

  // layer 1: 16x256, K=32 (padded)
  {
    v16h a = load_a16(feats, 32);
#pragma unroll
    for (int t = 0; t < 4; ++t) {
      int nt = w * 4 + t;
      v8f c = wmma16(a, load_bp(pW1, 16, 0, nt), zero8());
      int col = nt * 16 + ncol;
      float bb = b1[col];
#pragma unroll
      for (int j = 0; j < 8; ++j)
        H[(mb + j) * 256 + col] = (half_t)fast_tanh(c[j] + bb);
    }
  }
  __syncthreads();

  // layer 2: 16x128, K=256; filter * data -> scatter
#pragma unroll
  for (int t = 0; t < 2; ++t) {
    int nt = w * 2 + t;
    v8f acc = zero8();
#pragma unroll
    for (int kt = 0; kt < 8; ++kt)
      acc = wmma16(load_a16(H + kt * 32, 256), load_bp(pW2, 8, kt, nt), acc);
    int col = nt * 16 + ncol;
    float bb = b2[col];
    float we0 = W_eedata[col], we1 = W_eedata[128 + col];
    float we2 = W_eedata[256 + col], we3 = W_eedata[384 + col];
    float be = b_eedata[col];
#pragma unroll
    for (int j = 0; j < 8; ++j) {
      int m = mb + j;
      float f = acc[j] + bb;
      float data = fast_tanh(dvec[m * 4 + 0] * we0 + dvec[m * 4 + 1] * we1 +
                             dvec[m * 4 + 2] * we2 + dvec[m * 4 + 3] * we3 + be);
      atomic_add_f32(&e_emb[seg[m] * 128 + col], f * data);
    }
  }
}

// ---------------------------------------------------------------- norms

__global__ void en_norm_kernel(const float* __restrict__ en_dists,
                               const int* __restrict__ elec_idx,
                               const int* __restrict__ nuc_idx,
                               const float* __restrict__ charges,
                               float* __restrict__ e_norm) {
  int idx = blockIdx.x * 256 + threadIdx.x;
  if (idx >= N_EN) return;
  float v = __expf(-en_dists[idx * 4 + 3]) * charges[nuc_idx[idx]];
  atomic_add_f32(&e_norm[elec_idx[idx]], v);
}

__global__ void nn_norm_kernel(const float* __restrict__ nn_dists,
                               const int* __restrict__ nn_i,
                               const int* __restrict__ nn_j,
                               const float* __restrict__ charges,
                               float* __restrict__ n_neigh) {
  int idx = blockIdx.x * 256 + threadIdx.x;
  if (idx >= N_NN) return;
  float v = __expf(-nn_dists[idx * 4 + 3]) * charges[nn_j[idx]];
  atomic_add_f32(&n_neigh[nn_i[idx]], v);
}

__global__ void elecA_kernel(const float* __restrict__ e_emb,
                             const float* __restrict__ e_norm,
                             float* __restrict__ elecA) {
  int idx = blockIdx.x * 256 + threadIdx.x;  // 4096*256
  int i = idx >> 8, ch = idx & 255;
  float v = (ch < 128) ? e_emb[i * 128 + ch]
                       : e_emb[(N_ELEC + i) * 128 + (ch - 128)];
  elecA[idx] = v / (e_norm[i] + 1.0f);
}

// ---------------------------------------------------------------- en edges

__global__ __launch_bounds__(256) void en_kernel(
    const float* __restrict__ en_dists, const int* __restrict__ elec_idx,
    const int* __restrict__ nuc_idx, const int* __restrict__ spin_mask,
    const float* __restrict__ nuc_kernel, const float* __restrict__ nuc_bias,
    const float* __restrict__ elecA, const half_t* __restrict__ pWen1,
    const half_t* __restrict__ pWen2, const float* __restrict__ b_en1,
    const float* __restrict__ b_en2, const half_t* __restrict__ pWscale,
    float* __restrict__ agg_elec, float* __restrict__ agg_nuc,
    half_t* __restrict__ en_edge_out) {
  __shared__ __align__(16) half_t feats[32 * 32];
  __shared__ float ld4[32 * 4];
  __shared__ int eidx[32], nidx[32], spn[32];
  __shared__ __align__(16) half_t H[32 * 256];
  __shared__ __align__(16) half_t EE[32 * 128];
  __shared__ float EM[32 * 256];

  const int tid = threadIdx.x;
  const int base = blockIdx.x * 32;

  if (tid < 32) {
    int e = base + tid;
    int ei = elec_idx[e];
    eidx[tid] = ei;
    nidx[tid] = nuc_idx[e];
    spn[tid] = spin_mask[ei];
  }
  if (tid < 128) {
    int m = tid >> 2, c = tid & 3;
    int e = base + m;
    float r = en_dists[e * 4 + 3];
    ld4[tid] = en_dists[e * 4 + c] * (log1pf(r) / r);
  }
#pragma unroll
  for (int i = 0; i < 4; ++i) {
    int s = tid + i * 256;
    int m = s >> 5, k = s & 31;
    float r = en_dists[(base + m) * 4 + 3];
    feats[s] = (k < 16) ? (half_t)__expf(-r * (float)(k + 1)) : (half_t)0.f;
  }
  __syncthreads();

  // en_emb = tanh(ld4 . nuc_kernel + nuc_bias + elecA[elec])
  {
    const int ch = tid;
    for (int m = 0; m < 32; ++m) {
      int nuc = nidx[m];
      float acc = nuc_bias[nuc * EMB + ch];
#pragma unroll
      for (int c = 0; c < 4; ++c)
        acc += ld4[m * 4 + c] * nuc_kernel[(nuc * 4 + c) * EMB + ch];
      acc += elecA[eidx[m] * EMB + ch];
      EM[m * EMB + ch] = fast_tanh(acc);
    }
  }
  __syncthreads();

  const int w = tid >> 5, lane = tid & 31;
  const int g = w >> 2, ww = w & 3;
  const int ncol = lane & 15, mb = (lane >> 4) << 3;

  // edge MLP layer 1
  {
    v16h a = load_a16(feats + g * 16 * 32, 32);
#pragma unroll
    for (int t = 0; t < 4; ++t) {
      int nt = ww * 4 + t;
      v8f c = wmma16(a, load_bp(pWen1, 16, 0, nt), zero8());
      int col = nt * 16 + ncol;
      float bb = b_en1[col];
#pragma unroll
      for (int j = 0; j < 8; ++j)
        H[(g * 16 + mb + j) * 256 + col] = (half_t)fast_tanh(c[j] + bb);
    }
  }
  __syncthreads();

  // edge MLP layer 2 -> en_edge (LDS + global stash)
#pragma unroll
  for (int t = 0; t < 2; ++t) {
    int nt = ww * 2 + t;
    v8f acc = zero8();
#pragma unroll
    for (int kt = 0; kt < 8; ++kt)
      acc = wmma16(load_a16(H + g * 16 * 256 + kt * 32, 256),
                   load_bp(pWen2, 8, kt, nt), acc);
    int col = nt * 16 + ncol;
    float bb = b_en2[col];
#pragma unroll
    for (int j = 0; j < 8; ++j) {
      int m = g * 16 + mb + j;
      half_t v = (half_t)(acc[j] + bb);
      EE[m * 128 + col] = v;
      en_edge_out[(size_t)(base + m) * 128 + col] = v;
    }
  }
  __syncthreads();

  // scale = en_edge @ W_scale (N=512); messages to electrons (cols 0-255)
  // and nuclei (cols 256-511, spin-split)
#pragma unroll
  for (int t = 0; t < 8; ++t) {
    int nt = ww * 8 + t;
    v8f acc = zero8();
#pragma unroll
    for (int kt = 0; kt < 4; ++kt)
      acc = wmma16(load_a16(EE + g * 16 * 128 + kt * 32, 128),
                   load_bp(pWscale, 32, kt, nt), acc);
    int ch = (nt & 15) * 16 + ncol;
    if (nt < 16) {
      float run = 0.f;
      int cur = -1;
#pragma unroll
      for (int j = 0; j < 8; ++j) {
        int m = g * 16 + mb + j;
        float v = acc[j] * EM[m * EMB + ch];
        int e = eidx[m];
        if (e != cur) {
          if (cur >= 0) atomic_add_f32(&agg_elec[cur * EMB + ch], run);
          run = v;
          cur = e;
        } else {
          run += v;
        }
      }
      atomic_add_f32(&agg_elec[cur * EMB + ch], run);
    } else {
#pragma unroll
      for (int j = 0; j < 8; ++j) {
        int m = g * 16 + mb + j;
        float v = acc[j] * EM[m * EMB + ch];
        atomic_add_f32(&agg_nuc[((spn[m] ? 512 : 0) + nidx[m]) * EMB + ch], v);
      }
    }
  }
}

// ---------------------------------------------------------------- nuclei

__global__ void nuc_norm_kernel(const float* __restrict__ agg_nuc,
                                const float* __restrict__ n_neigh,
                                float* __restrict__ X, half_t* __restrict__ Xh) {
  int idx = blockIdx.x * 256 + threadIdx.x;  // 1024*256
  int r = idx >> 8;
  float v = agg_nuc[idx] / (n_neigh[r & 511] + 1.0f);
  X[idx] = v;
  Xh[idx] = (half_t)v;
}

__global__ __launch_bounds__(128) void nuc_update_kernel(
    const float* __restrict__ Xin, const half_t* __restrict__ Xinh,
    const half_t* __restrict__ pWs, const half_t* __restrict__ pWd,
    const float* __restrict__ bs, const float* __restrict__ bd,
    const float* __restrict__ bn, float* __restrict__ Xout,
    half_t* __restrict__ Xouth) {
  const int tid = threadIdx.x, w = tid >> 5, lane = tid & 31;
  const int tile = blockIdx.x * 4 + w;       // 1024 tiles = 64 x 16
  const int tm = tile >> 4, tn = tile & 15;
  const int ncol = lane & 15, mb = (lane >> 4) << 3;
  const half_t* aS = Xinh + tm * 16 * 256;
  const half_t* aD = Xinh + ((tm * 16) ^ 512) * 256;  // up<->down swap
  v8f acc = zero8();
#pragma unroll
  for (int kt = 0; kt < 8; ++kt) {
    acc = wmma16(load_a16(aS + kt * 32, 256), load_bp(pWs, 16, kt, tn), acc);
    acc = wmma16(load_a16(aD + kt * 32, 256), load_bp(pWd, 16, kt, tn), acc);
  }
  int col = tn * 16 + ncol;
  float bsum = bs[col] + bd[col];
#pragma unroll
  for (int j = 0; j < 8; ++j) {
    int r = tm * 16 + mb + j;
    float y = (acc[j] + bsum) * INV_SQRT2 + bn[(r & 511) * 256 + col];
    float v = (Xin[r * 256 + col] + fast_tanh(y)) * INV_SQRT2;
    Xout[r * 256 + col] = v;
    Xouth[r * 256 + col] = (half_t)v;
  }
}

// ---------------------------------------------------------------- final edges

__global__ __launch_bounds__(256) void final_edge_kernel(
    const half_t* __restrict__ en_edge, const half_t* __restrict__ pWw,
    const float* __restrict__ X, const int* __restrict__ elec_idx,
    const int* __restrict__ nuc_idx, const int* __restrict__ spin_mask,
    float* __restrict__ sfinal) {
  __shared__ __align__(16) half_t EE[32 * 128];
  __shared__ int eidx[32], nidx[32], spn[32];
  const int tid = threadIdx.x;
  const int base = blockIdx.x * 32;
  if (tid < 32) {
    int e = base + tid;
    int ei = elec_idx[e];
    eidx[tid] = ei;
    nidx[tid] = nuc_idx[e];
    spn[tid] = spin_mask[ei];
  }
#pragma unroll
  for (int i = 0; i < 16; ++i) {
    int s = tid + i * 256;
    EE[s] = en_edge[(size_t)base * 128 + s];
  }
  __syncthreads();

  const int w = tid >> 5, lane = tid & 31;
  const int g = w >> 2, ww = w & 3;
  const int ncol = lane & 15, mb = (lane >> 4) << 3;

#pragma unroll
  for (int t = 0; t < 4; ++t) {
    int nt = ww * 4 + t;
    v8f acc = zero8();
#pragma unroll
    for (int kt = 0; kt < 4; ++kt)
      acc = wmma16(load_a16(EE + g * 16 * 128 + kt * 32, 128),
                   load_bp(pWw, 16, kt, nt), acc);
    int ch = nt * 16 + ncol;
    float run = 0.f;
    int cur = -1;
#pragma unroll
    for (int j = 0; j < 8; ++j) {
      int m = g * 16 + mb + j;
      // spin true -> nuc_up (rows 0..511), false -> nuc_down (512..1023)
      float nv = X[((spn[m] ? 0 : 512) + nidx[m]) * 256 + ch];
      float v = acc[j] * nv;
      int e = eidx[m];
      if (e != cur) {
        if (cur >= 0) atomic_add_f32(&sfinal[cur * 256 + ch], run);
        run = v;
        cur = e;
      } else {
        run += v;
      }
    }
    atomic_add_f32(&sfinal[cur * 256 + ch], run);
  }
}

// ---------------------------------------------------------------- outputs

__global__ void elecB_kernel(const float* __restrict__ agg_elec,
                             const float* __restrict__ e_norm,
                             float* __restrict__ elecB,
                             half_t* __restrict__ elecBh) {
  int idx = blockIdx.x * 256 + threadIdx.x;  // 4096*256
  int i = idx >> 8;
  float v = agg_elec[idx] / (e_norm[i] + 1.0f);
  elecB[idx] = v;
  elecBh[idx] = (half_t)v;
}

__global__ __launch_bounds__(128) void out1_kernel(
    const half_t* __restrict__ elecBh, const half_t* __restrict__ pWd1,
    const float* __restrict__ bd1, const float* __restrict__ sfinal,
    const float* __restrict__ e_norm, half_t* __restrict__ Th) {
  const int tid = threadIdx.x, w = tid >> 5, lane = tid & 31;
  const int tile = blockIdx.x * 4 + w;       // 4096 tiles = 256 x 16
  const int tm = tile >> 4, tn = tile & 15;
  const int ncol = lane & 15, mb = (lane >> 4) << 3;
  v8f acc = zero8();
#pragma unroll
  for (int kt = 0; kt < 8; ++kt)
    acc = wmma16(load_a16(elecBh + tm * 16 * 256 + kt * 32, 256),
                 load_bp(pWd1, 16, kt, tn), acc);
  int col = tn * 16 + ncol;
  float bb = bd1[col];
#pragma unroll
  for (int j = 0; j < 8; ++j) {
    int i = tm * 16 + mb + j;
    float gate = sfinal[i * 256 + col] / (e_norm[i] + 1.0f);
    Th[i * 256 + col] = (half_t)fast_tanh((acc[j] + bb) * gate);
  }
}

__global__ __launch_bounds__(128) void out2_kernel(
    const half_t* __restrict__ Th, const half_t* __restrict__ pWd2,
    const float* __restrict__ bd2, const float* __restrict__ elecB,
    float* __restrict__ out) {
  const int tid = threadIdx.x, w = tid >> 5, lane = tid & 31;
  const int tile = blockIdx.x * 4 + w;
  const int tm = tile >> 4, tn = tile & 15;
  const int ncol = lane & 15, mb = (lane >> 4) << 3;
  v8f acc = zero8();
#pragma unroll
  for (int kt = 0; kt < 8; ++kt)
    acc = wmma16(load_a16(Th + tm * 16 * 256 + kt * 32, 256),
                 load_bp(pWd2, 16, kt, tn), acc);
  int col = tn * 16 + ncol;
  float bb = bd2[col];
#pragma unroll
  for (int j = 0; j < 8; ++j) {
    int i = tm * 16 + mb + j;
    float v = fast_tanh(acc[j] + bb);
    out[i * 256 + col] = (elecB[i * 256 + col] + v) * INV_SQRT2;
  }
}

// ---------------------------------------------------------------- launch

extern "C" void kernel_launch(void* const* d_in, const int* in_sizes, int n_in,
                              void* d_out, int out_size, void* d_ws,
                              size_t ws_size, hipStream_t stream) {
  const float* ee_dists = (const float*)d_in[0];
  const float* en_dists = (const float*)d_in[1];
  const float* nn_dists = (const float*)d_in[2];
  const float* charges = (const float*)d_in[3];
  const float* W_same1 = (const float*)d_in[4];
  const float* b_same1 = (const float*)d_in[5];
  const float* W_same2 = (const float*)d_in[6];
  const float* b_same2 = (const float*)d_in[7];
  const float* W_diff1 = (const float*)d_in[8];
  const float* b_diff1 = (const float*)d_in[9];
  const float* W_diff2 = (const float*)d_in[10];
  const float* b_diff2 = (const float*)d_in[11];
  const float* W_eedata = (const float*)d_in[12];
  const float* b_eedata = (const float*)d_in[13];
  const float* nuc_kernel = (const float*)d_in[14];
  const float* nuc_bias = (const float*)d_in[15];
  const float* W_en1 = (const float*)d_in[16];
  const float* b_en1 = (const float*)d_in[17];
  const float* W_en2 = (const float*)d_in[18];
  const float* b_en2 = (const float*)d_in[19];
  const float* W_scale = (const float*)d_in[20];
  const float* u_Ws[2] = {(const float*)d_in[21], (const float*)d_in[26]};
  const float* u_bs[2] = {(const float*)d_in[22], (const float*)d_in[27]};
  const float* u_Wd[2] = {(const float*)d_in[23], (const float*)d_in[28]};
  const float* u_bd[2] = {(const float*)d_in[24], (const float*)d_in[29]};
  const float* u_bn[2] = {(const float*)d_in[25], (const float*)d_in[30]};
  const float* Wd1 = (const float*)d_in[31];
  const float* bd1 = (const float*)d_in[32];
  const float* Ww = (const float*)d_in[33];
  const float* Wd2 = (const float*)d_in[34];
  const float* bd2 = (const float*)d_in[35];
  const int* elec_idx = (const int*)d_in[36];
  const int* nuc_idx = (const int*)d_in[37];
  const int* ee_i = (const int*)d_in[38];
  const int* nn_i = (const int*)d_in[39];
  const int* nn_j = (const int*)d_in[40];
  const int* spin = (const int*)d_in[41];
  (void)in_sizes; (void)n_in; (void)out_size; (void)ws_size;

  char* ws = (char*)d_ws;
  size_t o = 0;
  auto alloc = [&](size_t bytes) -> void* {
    void* p = ws + o;
    o += (bytes + 255) & ~(size_t)255;
    return p;
  };
  half_t* pW1s = (half_t*)alloc(8192 * 2);
  half_t* pW1d = (half_t*)alloc(8192 * 2);
  half_t* pWen1 = (half_t*)alloc(8192 * 2);
  half_t* pW2s = (half_t*)alloc(32768 * 2);
  half_t* pW2d = (half_t*)alloc(32768 * 2);
  half_t* pWen2 = (half_t*)alloc(32768 * 2);
  half_t* pWscale = (half_t*)alloc(65536 * 2);
  half_t* pWw = (half_t*)alloc(32768 * 2);
  half_t* pWd1 = (half_t*)alloc(65536 * 2);
  half_t* pWd2 = (half_t*)alloc(65536 * 2);
  half_t* pUws0 = (half_t*)alloc(65536 * 2);
  half_t* pUwd0 = (half_t*)alloc(65536 * 2);
  half_t* pUws1 = (half_t*)alloc(65536 * 2);
  half_t* pUwd1 = (half_t*)alloc(65536 * 2);
  float* e_emb = (float*)alloc((size_t)8192 * 128 * 4);
  float* e_norm = (float*)alloc(4096 * 4);
  float* n_neigh = (float*)alloc(512 * 4);
  float* elecA = (float*)alloc((size_t)4096 * 256 * 4);
  half_t* en_edge = (half_t*)alloc((size_t)131072 * 128 * 2);
  float* agg_elec = (float*)alloc((size_t)4096 * 256 * 4);
  float* agg_nuc = (float*)alloc((size_t)1024 * 256 * 4);
  float* XA = (float*)alloc((size_t)1024 * 256 * 4);
  half_t* XAh = (half_t*)alloc((size_t)1024 * 256 * 2);
  float* XB = (float*)alloc((size_t)1024 * 256 * 4);
  half_t* XBh = (half_t*)alloc((size_t)1024 * 256 * 2);
  float* sfinal = (float*)alloc((size_t)4096 * 256 * 4);
  float* elecB = (float*)alloc((size_t)4096 * 256 * 4);
  half_t* elecBh = (half_t*)alloc((size_t)4096 * 256 * 2);
  half_t* Th = (half_t*)alloc((size_t)4096 * 256 * 2);

  hipMemsetAsync(e_emb, 0, (size_t)8192 * 128 * 4, stream);
  hipMemsetAsync(e_norm, 0, 4096 * 4, stream);
  hipMemsetAsync(n_neigh, 0, 512 * 4, stream);
  hipMemsetAsync(agg_elec, 0, (size_t)4096 * 256 * 4, stream);
  hipMemsetAsync(agg_nuc, 0, (size_t)1024 * 256 * 4, stream);
  hipMemsetAsync(sfinal, 0, (size_t)4096 * 256 * 4, stream);

  auto packW = [&](const float* W, half_t* outp, int K, int N) {
    int KT = (K + 31) / 32;
    int total = KT * (N >> 4) * 512;
    pack_b_kernel<<<(total + 255) / 256, 256, 0, stream>>>(W, outp, K, N, total);
  };
  packW(W_same1, pW1s, 16, 256);
  packW(W_diff1, pW1d, 16, 256);
  packW(W_en1, pWen1, 16, 256);
  packW(W_same2, pW2s, 256, 128);
  packW(W_diff2, pW2d, 256, 128);
  packW(W_en2, pWen2, 256, 128);
  packW(W_scale, pWscale, 128, 512);
  packW(Ww, pWw, 128, 256);
  packW(Wd1, pWd1, 256, 256);
  packW(Wd2, pWd2, 256, 256);
  packW(u_Ws[0], pUws0, 256, 256);
  packW(u_Wd[0], pUwd0, 256, 256);
  packW(u_Ws[1], pUws1, 256, 256);
  packW(u_Wd[1], pUwd1, 256, 256);

  ee_kernel<<<8192, 128, 0, stream>>>(ee_dists, ee_i, pW1s, pW1d, pW2s, pW2d,
                                      b_same1, b_diff1, b_same2, b_diff2,
                                      W_eedata, b_eedata, e_emb);
  en_norm_kernel<<<512, 256, 0, stream>>>(en_dists, elec_idx, nuc_idx, charges,
                                          e_norm);
  nn_norm_kernel<<<64, 256, 0, stream>>>(nn_dists, nn_i, nn_j, charges, n_neigh);
  elecA_kernel<<<4096, 256, 0, stream>>>(e_emb, e_norm, elecA);
  en_kernel<<<4096, 256, 0, stream>>>(en_dists, elec_idx, nuc_idx, spin,
                                      nuc_kernel, nuc_bias, elecA, pWen1, pWen2,
                                      b_en1, b_en2, pWscale, agg_elec, agg_nuc,
                                      en_edge);
  nuc_norm_kernel<<<1024, 256, 0, stream>>>(agg_nuc, n_neigh, XA, XAh);
  nuc_update_kernel<<<256, 128, 0, stream>>>(XA, XAh, pUws0, pUwd0, u_bs[0],
                                             u_bd[0], u_bn[0], XB, XBh);
  nuc_update_kernel<<<256, 128, 0, stream>>>(XB, XBh, pUws1, pUwd1, u_bs[1],
                                             u_bd[1], u_bn[1], XA, XAh);
  final_edge_kernel<<<4096, 256, 0, stream>>>(en_edge, pWw, XA, elec_idx,
                                              nuc_idx, spin, sfinal);
  elecB_kernel<<<4096, 256, 0, stream>>>(agg_elec, e_norm, elecB, elecBh);
  out1_kernel<<<1024, 128, 0, stream>>>(elecBh, pWd1, bd1, sfinal, e_norm, Th);
  out2_kernel<<<1024, 128, 0, stream>>>(Th, pWd2, bd2, elecB, (float*)d_out);
}
